// LinearAttention_72043781423337
// MI455X (gfx1250) — matrix operand order repaired and checked by reading, not compiled, not run on hardware
//
#include <hip/hip_runtime.h>
#include <hip/hip_bf16.h>
#include <math.h>

// ---------------------------------------------------------------------------
// LinearAttention block for MI455X (gfx1250): bf16 WMMA everywhere.
// B=4, S=8192, D=512, H=8, hd=64.
// ---------------------------------------------------------------------------

#define DM    512
#define SLEN  8192
#define BATCH 4
#define NH    8
#define HD    64
#define LN_EPS 1e-5f

typedef __attribute__((ext_vector_type(16))) __bf16 v16bf;
typedef __attribute__((ext_vector_type(8)))  __bf16 bf16x8;
typedef __attribute__((ext_vector_type(8)))  float  v8f;

static __device__ __forceinline__ v8f wmma_bf16(v16bf a, v16bf b, v8f c) {
  return __builtin_amdgcn_wmma_f32_16x16x32_bf16(false, a, false, b,
                                                 (short)0, c, false, false);
}

// A-fragment (16x32 bf16) from a row-major bf16 tile.
// lane = (lrow, half): row = lrow, K chunks at half*8 and 16+half*8.
static __device__ __forceinline__ v16bf load_a_frag(const __bf16* base) {
  v16bf a;
#pragma unroll
  for (int e = 0; e < 8; ++e) { a[e] = base[e]; a[8 + e] = base[16 + e]; }
  return a;
}

// elu(x)+1  ==  x>0 ? x+1 : exp(x)
static __device__ __forceinline__ float feature_map(float x) {
  return x > 0.f ? x + 1.f : __expf(x);
}

// ---------------------------------------------------------------------------
// Kernel 1: pack W (f32, row-major KxN = 512x512) into WMMA-B fragment order:
//   packed[((kt*32 + nt)*32 + lane)*16 + e] = W[(kt*32 + half*8 + (e&7) + (e>>3)*16)*512 + nt*16 + (lane&15)]
// so a GEMM wave loads its B fragment as one contiguous 32-byte chunk.
// ---------------------------------------------------------------------------
__global__ void pack_w_kernel(const float* __restrict__ Wq, const float* __restrict__ Wk,
                              const float* __restrict__ Wv, const float* __restrict__ Wo,
                              __bf16* __restrict__ Pq, __bf16* __restrict__ Pk,
                              __bf16* __restrict__ Pv, __bf16* __restrict__ Po) {
  long idx = (long)blockIdx.x * blockDim.x + threadIdx.x;   // 4 * 262144 total
  int  mat = (int)(idx >> 18);
  int  p   = (int)(idx & 262143);
  int  e    = p & 15;
  int  lane = (p >> 4) & 31;
  int  nt   = (p >> 9) & 31;
  int  kt   = p >> 14;
  int  N = nt * 16 + (lane & 15);
  int  K = kt * 32 + (lane >> 4) * 8 + (e & 7) + (e >> 3) * 16;
  const float* src = (mat == 0) ? Wq : (mat == 1) ? Wk : (mat == 2) ? Wv : Wo;
  __bf16*      dst = (mat == 0) ? Pq : (mat == 1) ? Pk : (mat == 2) ? Pv : Po;
  dst[p] = (__bf16)src[K * DM + N];
}

// ---------------------------------------------------------------------------
// Kernel 2: zero kv (B*H*64*64 f32) and k_sum (B*H*64 f32) accumulators.
// ---------------------------------------------------------------------------
__global__ void zero_kv_kernel(float* __restrict__ kv, float* __restrict__ ksum) {
  int i = blockIdx.x * blockDim.x + threadIdx.x;
  if (i < BATCH * NH * HD * HD) kv[i] = 0.f;
  if (i < BATCH * NH * HD)      ksum[i] = 0.f;
}

// ---------------------------------------------------------------------------
// Kernel 3: fused QKV GEMM + bias + feature map.  M-blocked for B reuse:
// Grid: (B*S)/32 blocks of 384 threads (12 waves). Waves 0-3 -> Q, 4-7 -> K,
// 8-11 -> V; each wave owns 8 column tiles of ONE packed weight matrix and
// TWO 16-row tiles, so every loaded B fragment feeds two WMMAs (halves the
// weight stream). All B addresses are one base pointer + immediate offsets.
// launch_bounds + waves_per_eu(1): ~176 VGPRs live (16 acc tiles resident),
// trading occupancy (~5 waves/SIMD) for a spill-free inner loop.
// Q stored row-major bf16 (B*S, 512); K,V stored TRANSPOSED per head:
// KT/VT[((b*H+h)*64 + d)*S + s] so stage-4 fragments are contiguous.
// ---------------------------------------------------------------------------
__global__ __launch_bounds__(384, 1)
__attribute__((amdgpu_waves_per_eu(1)))
void qkv_kernel(const float* __restrict__ x,
                const __bf16* __restrict__ Pq, const __bf16* __restrict__ Pk,
                const __bf16* __restrict__ Pv,
                const float* __restrict__ bq, const float* __restrict__ bk,
                const float* __restrict__ bv,
                __bf16* __restrict__ Qf, __bf16* __restrict__ KT,
                __bf16* __restrict__ VT) {
  __shared__ __bf16 xs[32 * DM];                 // 32 KB staged x tile (bf16)
  const long row0 = (long)blockIdx.x * 32;       // flattened (b*S + s)
  const int  b    = (int)(row0 / SLEN);
  const int  s0   = (int)(row0 % SLEN);
  const int  tid  = threadIdx.x;

  for (int i = tid; i < 32 * DM; i += 384)
    xs[i] = (__bf16)x[row0 * DM + i];
  __syncthreads();

  const int wave = tid >> 5, lane = tid & 31;
  const int lrow = lane & 15, half = lane >> 4;
  const int mat  = wave >> 2;                    // 0=Q, 1=K, 2=V
  const int nt0  = (wave & 3) * 8;               // first of 8 column tiles

  const __bf16* Wp = (mat == 0) ? Pq : (mat == 1) ? Pk : Pv;
  // element index = kt*16384 + nt*512 + lane*16; hoist lane & nt0 terms.
  const __bf16* wbase = Wp + nt0 * 512 + lane * 16;
  const __bf16* arow0 = &xs[lrow * DM + half * 8];          // rows 0..15
  const __bf16* arow1 = arow0 + 16 * DM;                    // rows 16..31

  v8f acc0[8] = {};
  v8f acc1[8] = {};

#pragma unroll 1
  for (int kt = 0; kt < 16; ++kt) {
    v16bf a0 = load_a_frag(arow0 + kt * 32);
    v16bf a1 = load_a_frag(arow1 + kt * 32);
    const __bf16* wk = wbase + kt * 16384;
#pragma unroll
    for (int t = 0; t < 8; ++t) {
      v16bf bfr = *(const v16bf*)(wk + t * 512); // byte offset t*1024 (immediate)
      acc0[t] = wmma_bf16(a0, bfr, acc0[t]);     // B reused by both row tiles
      acc1[t] = wmma_bf16(a1, bfr, acc1[t]);
    }
  }

#pragma unroll 1
  for (int t = 0; t < 8; ++t) {
    const int nt  = nt0 + t;
    const int col = nt * 16 + lrow;              // N per lane
    if (mat == 0) {                              // Q: row-major bf16 store
      const float bias = bq[col];
#pragma unroll
      for (int rt = 0; rt < 2; ++rt) {
        const v8f* a = rt ? &acc1[t] : &acc0[t];
#pragma unroll
        for (int r = 0; r < 8; ++r) {
          const int m = rt * 16 + r + 8 * half;
          Qf[(row0 + m) * DM + col] = (__bf16)feature_map((*a)[r] + bias);
        }
      }
    } else {                                     // K or V: transposed store
      const int   h    = col >> 6, d = col & 63;
      const float bias = (mat == 1) ? bk[col] : bv[col];
      __bf16* base = ((mat == 1) ? KT : VT) +
                     ((long)((b * NH + h) * HD + d)) * SLEN + s0 + 8 * half;
#pragma unroll
      for (int rt = 0; rt < 2; ++rt) {
        const v8f* a = rt ? &acc1[t] : &acc0[t];
        bf16x8 tmp;
#pragma unroll
        for (int r = 0; r < 8; ++r) {
          float f = (*a)[r] + bias;
          if (mat == 1) f = feature_map(f);
          tmp[r] = (__bf16)f;
        }
        *(bf16x8*)(base + rt * 16) = tmp;        // 16B contiguous in s
      }
    }
  }
}

// ---------------------------------------------------------------------------
// Kernel 4: kv summary  kv[b,h,d,m] = sum_n K[n,d] * V[n,m]  and k_sum.
// Grid: (B*H)*16 chunks of 512 sequence positions; 128 threads (4 waves).
// Wave w owns d-tile w (16 d rows x 64 m cols = 4 WMMA tiles), reduced into
// the global kv accumulator with global_atomic_add_f32.
// ---------------------------------------------------------------------------
__global__ void kv_sum_kernel(const __bf16* __restrict__ KT, const __bf16* __restrict__ VT,
                              float* __restrict__ kv, float* __restrict__ ksum) {
  const int NCH = 16, CHUNK = SLEN / NCH;
  const int bh = blockIdx.x / NCH;
  const int n_base = (blockIdx.x % NCH) * CHUNK;
  const int tid = threadIdx.x;

  // k_sum partial reduction (threads 0..63 each own one d)
  if (tid < HD) {
    const __bf16* p = KT + ((long)(bh * HD + tid)) * SLEN + n_base;
    float s = 0.f;
    for (int n = 0; n < CHUNK; ++n) s += (float)p[n];
    atomicAdd(&ksum[bh * HD + tid], s);
  }

  const int wave = tid >> 5, lane = tid & 31;
  const int lrow = lane & 15, half = lane >> 4;
  const int dt = wave;

  const __bf16* arow = KT + ((long)(bh * HD + dt * 16 + lrow)) * SLEN
                          + n_base + half * 8;
  const __bf16* brow = VT + ((long)(bh * HD + lrow)) * SLEN
                          + n_base + half * 8;              // + mt*16 rows

  v8f acc[4] = {};

#pragma unroll 1
  for (int i = 0; i < CHUNK / 32; ++i) {
    v16bf a = load_a_frag(arow + i * 32);
#pragma unroll
    for (int mt = 0; mt < 4; ++mt) {
      v16bf bb;
      const __bf16* bp = brow + (long)(mt * 16) * SLEN + i * 32;
#pragma unroll
      for (int e = 0; e < 16; ++e) bb[e] = bp[(e & 7) + (e >> 3) * 16];
      acc[mt] = wmma_bf16(a, bb, acc[mt]);
    }
  }

#pragma unroll
  for (int mt = 0; mt < 4; ++mt)
#pragma unroll
    for (int r = 0; r < 8; ++r) {
      const int d = dt * 16 + r + 8 * half;
      atomicAdd(&kv[(bh * HD + d) * HD + mt * 16 + lrow], acc[mt][r]);
    }
}

// ---------------------------------------------------------------------------
// Kernel 5: pack kv (f32) into bf16 WMMA-B fragment order per (b,h):
//   kvp[bh*4096 + kt*2048 + nt*512 + lane*16 + e]
//     = kv[bh][kt*32 + (lane>>4)*8 + (e&7) + (e>>3)*16][nt*16 + (lane&15)]
// ---------------------------------------------------------------------------
__global__ void kv_pack_kernel(const float* __restrict__ kv, __bf16* __restrict__ kvp) {
  int idx = blockIdx.x * blockDim.x + threadIdx.x;   // B*H*64*64 = 131072
  int e    = idx & 15;
  int lane = (idx >> 4) & 31;
  int nt   = (idx >> 9) & 3;
  int kt   = (idx >> 11) & 1;
  int bh   = idx >> 12;
  int d = kt * 32 + (lane >> 4) * 8 + (e & 7) + (e >> 3) * 16;
  int m = nt * 16 + (lane & 15);
  kvp[idx] = (__bf16)kv[(bh * HD + d) * HD + m];
}

// ---------------------------------------------------------------------------
// Kernel 6: out = (Q @ kv) / max(Q . k_sum, 1e-6), heads merged -> bf16 attn.
// Grid: (B*S)/16 blocks, 256 threads (8 waves); wave w handles head w.
// ---------------------------------------------------------------------------
__global__ void attn_kernel(const __bf16* __restrict__ Qf, const __bf16* __restrict__ kvp,
                            const float* __restrict__ ksum, __bf16* __restrict__ attn) {
  __shared__ float norm_s[16][NH];
  const long row0 = (long)blockIdx.x * 16;
  const int  b    = (int)(row0 / SLEN);
  const int  tid  = threadIdx.x;

  if (tid < 16 * NH) {                           // normalizer q . k_sum
    const int sl = tid >> 3, h = tid & 7;
    const __bf16* qp = Qf + (row0 + sl) * DM + h * HD;
    const float*  kp = ksum + (b * NH + h) * HD;
    float dot = 0.f;
    for (int d = 0; d < HD; ++d) dot += (float)qp[d] * kp[d];
    norm_s[sl][h] = fmaxf(dot, 1e-6f);
  }
  __syncthreads();

  const int wave = tid >> 5, lane = tid & 31;
  const int lrow = lane & 15, half = lane >> 4;
  const int h  = wave;
  const int bh = b * NH + h;
  const __bf16* kb = kvp + bh * 4096 + lane * 16;

  v8f acc[4] = {};
#pragma unroll
  for (int kt = 0; kt < 2; ++kt) {
    v16bf a = load_a_frag(Qf + (row0 + lrow) * DM + h * HD + kt * 32 + half * 8);
    v16bf bb[4];
#pragma unroll
    for (int nt = 0; nt < 4; ++nt)
      bb[nt] = *(const v16bf*)(kb + kt * 2048 + nt * 512);
#pragma unroll
    for (int nt = 0; nt < 4; ++nt)
      acc[nt] = wmma_bf16(a, bb[nt], acc[nt]);
  }

#pragma unroll
  for (int nt = 0; nt < 4; ++nt)
#pragma unroll
    for (int r = 0; r < 8; ++r) {
      const int m = r + 8 * half;
      const float o = acc[nt][r] / norm_s[m][h];
      attn[(row0 + m) * DM + h * HD + nt * 16 + lrow] = (__bf16)o;
    }
}

// ---------------------------------------------------------------------------
// Kernel 7: y = attn @ Wo + bo + x, then LayerNorm * gamma + beta -> f32 out.
// Grid: (B*S)/16 blocks, 256 threads (8 waves); block owns a full 16x512 row
// slab so the LayerNorm can be fused with LDS reductions.
// ---------------------------------------------------------------------------
__global__ void out_ln_kernel(const __bf16* __restrict__ attn, const __bf16* __restrict__ Po,
                              const float* __restrict__ bo, const float* __restrict__ x,
                              const float* __restrict__ gamma, const float* __restrict__ beta,
                              float* __restrict__ out) {
  __shared__ __align__(16) char smem[16 * DM * 4];   // 32 KB, reused
  __bf16* as = (__bf16*)smem;                        // staged attn tile (16 KB)
  float*  yb = (float*)smem;                         // pre-LN y (32 KB, after barrier)
  __shared__ float red[16][16][2];
  __shared__ float stats[16][2];

  const long row0 = (long)blockIdx.x * 16;
  const int  tid  = threadIdx.x;

  for (int i = tid; i < 16 * DM; i += 256)
    as[i] = attn[row0 * DM + i];
  __syncthreads();

  const int wave = tid >> 5, lane = tid & 31;
  const int lrow = lane & 15, half = lane >> 4;
  const int nt0  = wave * 4;
  const __bf16* wbase = Po + nt0 * 512 + lane * 16;
  const __bf16* arow  = &as[lrow * DM + half * 8];

  v8f acc[4] = {};
#pragma unroll 1
  for (int kt = 0; kt < 16; ++kt) {
    v16bf a = load_a_frag(arow + kt * 32);
    const __bf16* wk = wbase + kt * 16384;
#pragma unroll
    for (int t = 0; t < 4; ++t) {
      v16bf bfr = *(const v16bf*)(wk + t * 512);
      acc[t] = wmma_bf16(a, bfr, acc[t]);
    }
  }
  __syncthreads();                               // done reading `as`; reuse as yb

  // bias + residual -> yb (pre-LN) in LDS
#pragma unroll
  for (int t = 0; t < 4; ++t) {
    const int col  = (nt0 + t) * 16 + lrow;
    const float bb = bo[col];
#pragma unroll
    for (int r = 0; r < 8; ++r) {
      const int m = r + 8 * half;
      yb[m * DM + col] = acc[t][r] + bb + x[(row0 + m) * DM + col];
    }
  }
  __syncthreads();

  // LayerNorm: per-row mean / variance via two-level LDS reduction
  {
    const int row = tid >> 4, sub = tid & 15;
    float s = 0.f, sq = 0.f;
    for (int j = 0; j < 32; ++j) {
      const float v = yb[row * DM + sub * 32 + j];
      s += v; sq += v * v;
    }
    red[row][sub][0] = s; red[row][sub][1] = sq;
  }
  __syncthreads();
  if (tid < 16) {
    float s = 0.f, sq = 0.f;
    for (int j = 0; j < 16; ++j) { s += red[tid][j][0]; sq += red[tid][j][1]; }
    const float mean = s * (1.f / DM);
    const float var  = sq * (1.f / DM) - mean * mean;
    stats[tid][0] = mean;
    stats[tid][1] = rsqrtf(var + LN_EPS);
  }
  __syncthreads();

  for (int i = tid; i < 16 * DM; i += 256) {
    const int row = i >> 9, c = i & (DM - 1);
    out[row0 * DM + i] = (yb[i] - stats[row][0]) * stats[row][1] * gamma[c] + beta[c];
  }
}

// ---------------------------------------------------------------------------
// Host launcher
// ---------------------------------------------------------------------------
extern "C" void kernel_launch(void* const* d_in, const int* in_sizes, int n_in,
                              void* d_out, int out_size, void* d_ws, size_t ws_size,
                              hipStream_t stream) {
  (void)in_sizes; (void)n_in; (void)out_size; (void)ws_size;

  const float* x     = (const float*)d_in[0];
  const float* Wq    = (const float*)d_in[1];
  const float* bq    = (const float*)d_in[2];
  const float* Wk    = (const float*)d_in[3];
  const float* bk    = (const float*)d_in[4];
  const float* Wv    = (const float*)d_in[5];
  const float* bv    = (const float*)d_in[6];
  const float* Wo    = (const float*)d_in[7];
  const float* bo    = (const float*)d_in[8];
  const float* gamma = (const float*)d_in[9];
  const float* beta  = (const float*)d_in[10];
  float* out = (float*)d_out;

  // workspace layout (bytes)
  char* ws = (char*)d_ws;
  const size_t SZ_PK  = (size_t)DM * DM * 2;            // 512 KB each
  const size_t SZ_BIG = (size_t)BATCH * SLEN * DM * 2;  // 32 MB each
  __bf16* Pq   = (__bf16*)(ws);
  __bf16* Pk   = (__bf16*)(ws + SZ_PK);
  __bf16* Pv   = (__bf16*)(ws + 2 * SZ_PK);
  __bf16* Po   = (__bf16*)(ws + 3 * SZ_PK);
  float*  kv   = (float*) (ws + 4 * SZ_PK);                             // 512 KB
  float*  ksum = (float*) (ws + 4 * SZ_PK + (size_t)BATCH*NH*HD*HD*4);  // 8 KB
  __bf16* kvp  = (__bf16*)(ws + 4 * SZ_PK + (size_t)BATCH*NH*HD*HD*4 + 32768); // 256 KB
  char*   big  = ws + 4 * SZ_PK + (size_t)BATCH*NH*HD*HD*4 + 32768 + (size_t)BATCH*NH*HD*HD*2 + 65536;
  __bf16* Qf   = (__bf16*)(big);
  __bf16* KT   = (__bf16*)(big + SZ_BIG);
  __bf16* VT   = (__bf16*)(big + 2 * SZ_BIG);
  __bf16* attn = (__bf16*)(big + 3 * SZ_BIG);

  pack_w_kernel<<<(4 * DM * DM) / 256, 256, 0, stream>>>(Wq, Wk, Wv, Wo, Pq, Pk, Pv, Po);
  zero_kv_kernel<<<(BATCH * NH * HD * HD + 255) / 256, 256, 0, stream>>>(kv, ksum);
  qkv_kernel<<<(BATCH * SLEN) / 32, 384, 0, stream>>>(x, Pq, Pk, Pv, bq, bk, bv, Qf, KT, VT);
  kv_sum_kernel<<<BATCH * NH * 16, 128, 0, stream>>>(KT, VT, kv, ksum);
  kv_pack_kernel<<<(BATCH * NH * HD * HD) / 256, 256, 0, stream>>>(kv, kvp);
  attn_kernel<<<(BATCH * SLEN) / 16, 256, 0, stream>>>(Qf, kvp, ksum, attn);
  out_ln_kernel<<<(BATCH * SLEN) / 16, 256, 0, stream>>>(attn, Po, bo, x, gamma, beta, out);
}
